// ExMRD_Retrieval_10557029613954
// MI455X (gfx1250) — compile-verified, matrix operand order checked by reading, not run on hardware
//
#include <hip/hip_runtime.h>
#include <hip/hip_bf16.h>
#include <math.h>

// ---------------- problem constants (from reference setup_inputs) ------------
constexpr int   Bq    = 1024;     // queries
constexpr int   Nmem  = 100000;   // memory rows
constexpr int   Ddim  = 256;      // feature dim
constexpr int   TOPK  = 16;
constexpr int   CHUNK = 4096;                         // N columns per block.y
constexpr int   NCHUNK = (Nmem + CHUNK - 1) / CHUNK;  // 25
constexpr float NEGF  = -3.0e38f;
constexpr float EPSF  = 1e-8f;
constexpr float TEMP  = 0.07f;

typedef __attribute__((ext_vector_type(16))) _Float16 v16h;
typedef __attribute__((ext_vector_type(8)))  _Float16 v8h;
typedef __attribute__((ext_vector_type(8)))  float    v8f;

#define SHUF16(lo, hi) __builtin_shufflevector((lo), (hi), \
        0,1,2,3,4,5,6,7,8,9,10,11,12,13,14,15)

// ---------------- kernel 1: row L2-normalize fp32 -> f16 ---------------------
__global__ __launch_bounds__(256)
void normalize_rows_f16(const float* __restrict__ src,
                        _Float16* __restrict__ dst,
                        int cols /* == 256 */) {
    __shared__ float red[256];
    const int row = blockIdx.x;
    const int c   = threadIdx.x;
    const float x = src[(size_t)row * cols + c];
    red[c] = x * x;
    __syncthreads();
    #pragma unroll
    for (int s = 128; s > 0; s >>= 1) {
        if (c < s) red[c] += red[c + s];
        __syncthreads();
    }
    const float nrm   = sqrtf(red[0]);
    const float scale = 1.0f / fmaxf(nrm, EPSF);
    dst[(size_t)row * cols + c] = (_Float16)(x * scale);
}

// ---------------- kernel 2: WMMA cosine-sim GEMM + fused chunk top-16 --------
// grid = (Bq/32, NCHUNK), block = 256 (8 waves).
// 32 queries per block: two A-fragment sets (128 VGPRs) live in registers for
// the whole column loop; every B fragment loaded from the L2-resident f16
// bank feeds TWO v_wmma ops (2x B reuse -> half the L2 traffic vs M=16).
// Each wave owns one 16-column tile per iteration; a 32x128 sims slab goes
// through LDS and 8 scan-threads per query maintain private top-16 lists.
__global__ __launch_bounds__(256)
void sim_topk_kernel(const _Float16* __restrict__ qn,
                     const _Float16* __restrict__ mn,
                     float* __restrict__ pV,
                     int*   __restrict__ pI) {
    __shared__ __align__(16) _Float16 Alds[32 * 256];       // 16 KB query tile
    __shared__ float Slds[32 * 128];                        // 16 KB sims slab
    __shared__ float MV[32 * 8 * TOPK];                     // 16 KB merge vals
    __shared__ int   MI[32 * 8 * TOPK];                     // 16 KB merge idx

    const int qt    = blockIdx.x;      // 32-query tile
    const int chunk = blockIdx.y;      // N chunk
    const int tid   = threadIdx.x;
    const int wave  = tid >> 5;
    const int lane  = tid & 31;

    // Stage A tile (32 x 256 f16) into LDS: 8 halves per transfer.
    {
        const v8h* src = (const v8h*)(qn + (size_t)qt * 32 * 256);
        v8h* dst = (v8h*)Alds;
        #pragma unroll
        for (int i = 0; i < 4; ++i) dst[tid + i * 256] = src[tid + i * 256];
    }
    __syncthreads();

    // Lane mapping per the 16-bit WMMA layouts (ISA 7.12.2):
    //  A 16x32: lanes 0-15 -> M=lane, K in {0..7,16..23}; lanes 16-31 -> K in {8..15,24..31}
    //  B 32x16: lanes 0-15 -> N=lane, same K split.
    const int nl   = lane & 15;
    const int mo   = (lane >> 4) ? 8 : 0;    // C-tile row offset for this lane half
    const int ksel = (lane >> 4) ? 8 : 0;    // K sub-offset within a 32-wide K step

    // Hoist A fragments for BOTH 16-query M-tiles (invariant across columns).
    v16h af0[8], af1[8];
    {
        const _Float16* arow0 = Alds + nl * 256;
        const _Float16* arow1 = Alds + (16 + nl) * 256;
        #pragma unroll
        for (int kk = 0; kk < 8; ++kk) {
            v8h a0l = *(const v8h*)(arow0 + kk * 32 + ksel);
            v8h a0h = *(const v8h*)(arow0 + kk * 32 + 16 + ksel);
            v8h a1l = *(const v8h*)(arow1 + kk * 32 + ksel);
            v8h a1h = *(const v8h*)(arow1 + kk * 32 + 16 + ksel);
            af0[kk] = SHUF16(a0l, a0h);
            af1[kk] = SHUF16(a1l, a1h);
        }
    }

    // Private top-16 state (per scan-thread).
    float tv[TOPK]; int tix[TOPK];
    #pragma unroll
    for (int i = 0; i < TOPK; ++i) { tv[i] = NEGF; tix[i] = 0; }
    float tmin = NEGF; int tminpos = 0;

    const int q   = tid >> 3;   // query row this thread scans (0..31)
    const int sub = tid & 7;    // column phase within slab

    for (int iter = 0; iter < CHUNK / 128; ++iter) {
        const int colBase = chunk * CHUNK + iter * 128;

        // ---- WMMA: this wave's 16-column tile against both M-tiles ----
        {
            const int c0 = colBase + wave * 16 + nl;
            const int n0 = c0 < Nmem ? c0 : (Nmem - 1);
            const _Float16* p0 = mn + (size_t)n0 * Ddim + ksel;

            // Prefetch next iteration's row (L2-resident bank -> hide latency).
            {
                int pf = c0 + 128; pf = pf < Nmem ? pf : (Nmem - 1);
                __builtin_prefetch(mn + (size_t)pf * Ddim, 0, 1);
            }

            v8f accA = {}, accB = {};
            #pragma unroll
            for (int kk = 0; kk < 8; ++kk) {
                v8h bl = *(const v8h*)(p0 + kk * 32);
                v8h bh = *(const v8h*)(p0 + kk * 32 + 16);
                const v16h bf = SHUF16(bl, bh);
                accA = __builtin_amdgcn_wmma_f32_16x16x32_f16(
                        false, af0[kk], false, bf, (short)0, accA, false, false);
                accB = __builtin_amdgcn_wmma_f32_16x16x32_f16(
                        false, af1[kk], false, bf, (short)0, accB, false, false);
            }

            const bool vld = c0 < Nmem;
            #pragma unroll
            for (int r = 0; r < 8; ++r) {
                Slds[(mo + r) * 128 + wave * 16 + nl]      = vld ? accA[r] : NEGF;
                Slds[(16 + mo + r) * 128 + wave * 16 + nl] = vld ? accB[r] : NEGF;
            }
        }
        __syncthreads();

        // ---- top-16 scan: 8 threads per query, 16 columns each ----
        #pragma unroll
        for (int cc = 0; cc < 16; ++cc) {
            const int c = sub + cc * 8;
            const float v = Slds[q * 128 + c];
            if (v > tmin) {
                tv[tminpos]  = v;
                tix[tminpos] = colBase + c;
                float m2 = tv[0]; int mp = 0;
                #pragma unroll
                for (int e = 1; e < TOPK; ++e)
                    if (tv[e] < m2) { m2 = tv[e]; mp = e; }
                tmin = m2; tminpos = mp;
            }
        }
        __syncthreads();
    }

    // Dump private lists for merge.
    #pragma unroll
    for (int e = 0; e < TOPK; ++e) {
        MV[(q * 8 + sub) * TOPK + e] = tv[e];
        MI[(q * 8 + sub) * TOPK + e] = tix[e];
    }
    __syncthreads();

    // Per-query merge of 128 candidates -> chunk-local top-16.
    if (tid < 32) {
        const int qq = tid;
        float* vals = MV + qq * 8 * TOPK;
        int*   idxs = MI + qq * 8 * TOPK;
        const size_t obase = ((size_t)(qt * 32 + qq) * NCHUNK + chunk) * TOPK;
        for (int s = 0; s < TOPK; ++s) {
            float best = NEGF; int bp = 0;
            for (int j = 0; j < 8 * TOPK; ++j)
                if (vals[j] > best) { best = vals[j]; bp = j; }
            pV[obase + s] = best;
            pI[obase + s] = idxs[bp];
            vals[bp] = NEGF;
        }
    }
}

// ---------------- kernel 3: global top-16, softmax, DST fusion ---------------
__device__ inline void a2bu(const float a0, const float a1,
                            float& b0, float& b1, float& u) {
    const float S = fmaxf(a0 + a1, EPSF);
    b0 = fmaxf((a0 - 1.f) / S, 0.f);
    b1 = fmaxf((a1 - 1.f) / S, 0.f);
    u  = fminf(fmaxf(2.f / S, EPSF), 1.f - EPSF);
    const float bs = b0 + b1;
    const float sc = fmaxf(1.f - u, EPSF) / fmaxf(bs, EPSF);
    b0 *= sc; b1 *= sc;
}

__global__ __launch_bounds__(256)
void finalize_kernel(const float* __restrict__ pV,
                     const int*   __restrict__ pI,
                     const float* __restrict__ mem_ev,
                     const float* __restrict__ model_ev,
                     float* __restrict__ out) {
    const int b = blockIdx.x * blockDim.x + threadIdx.x;
    if (b >= Bq) return;

    constexpr int NC = NCHUNK * TOPK;   // 400 candidates
    float cv[NC]; int ci[NC];
    const float* pv = pV + (size_t)b * NC;
    const int*   pi = pI + (size_t)b * NC;
    for (int j = 0; j < NC; ++j) { cv[j] = pv[j]; ci[j] = pi[j]; }

    float topv[TOPK]; int topi[TOPK];
    for (int s = 0; s < TOPK; ++s) {
        float best = NEGF; int bp = 0;
        for (int j = 0; j < NC; ++j)
            if (cv[j] > best) { best = cv[j]; bp = j; }
        topv[s] = best; topi[s] = ci[bp]; cv[bp] = NEGF;
    }

    // softmax(topv / T)
    const float vmax = topv[0];
    float w[TOPK]; float wsum = 0.f;
    #pragma unroll
    for (int s = 0; s < TOPK; ++s) {
        w[s] = __expf((topv[s] - vmax) / TEMP);
        wsum += w[s];
    }
    const float inv = 1.f / wsum;

    // retrieval opinion alpha_r = 1 + sum w * evidence
    float er0 = 0.f, er1 = 0.f;
    for (int s = 0; s < TOPK; ++s) {
        const size_t n = (size_t)topi[s];
        const float ws = w[s] * inv;
        er0 += ws * mem_ev[n * 2 + 0];
        er1 += ws * mem_ev[n * 2 + 1];
    }
    const float ar0 = 1.f + er0, ar1 = 1.f + er1;
    const float am0 = model_ev[(size_t)b * 2 + 0] + 1.f;
    const float am1 = model_ev[(size_t)b * 2 + 1] + 1.f;

    float bm0, bm1, um, br0, br1, ur;
    a2bu(am0, am1, bm0, bm1, um);
    a2bu(ar0, ar1, br0, br1, ur);

    // combine_two_opinions
    const float tp = (bm0 + bm1) * (br0 + br1);
    const float ds = bm0 * br0 + bm1 * br1;
    const float C  = tp - ds;
    const float S  = fmaxf(1.f - C, EPSF);
    float bf0 = (bm0 * br0 + bm0 * ur + br0 * um) / S;
    float bf1 = (bm1 * br1 + bm1 * ur + br1 * um) / S;
    float uf  = um * ur / S;
    bf0 = fmaxf(bf0, 0.f); bf1 = fmaxf(bf1, 0.f);
    uf  = fminf(fmaxf(uf, EPSF), 1.f - EPSF);
    const float bs = bf0 + bf1;
    const float sc = (1.f - uf) / fmaxf(bs, EPSF);
    bf0 *= sc; bf1 *= sc;

    // opinion_to_alpha
    uf = fminf(fmaxf(uf, EPSF), 1.f - EPSF);
    const float Sf = 2.f / uf;
    out[(size_t)b * 2 + 0] = fmaxf(bf0 * Sf + 1.f, 1.f + EPSF);
    out[(size_t)b * 2 + 1] = fmaxf(bf1 * Sf + 1.f, 1.f + EPSF);
}

// ---------------- launch ------------------------------------------------------
extern "C" void kernel_launch(void* const* d_in, const int* in_sizes, int n_in,
                              void* d_out, int out_size, void* d_ws, size_t ws_size,
                              hipStream_t stream) {
    const float* query  = (const float*)d_in[0];   // (1024, 256)
    const float* memf   = (const float*)d_in[1];   // (100000, 256)
    const float* memev  = (const float*)d_in[2];   // (100000, 2)
    const float* modev  = (const float*)d_in[3];   // (1024, 2)
    (void)in_sizes; (void)n_in; (void)out_size; (void)ws_size;

    // Workspace layout
    char* ws = (char*)d_ws;
    size_t off = 0;
    _Float16* qn = (_Float16*)(ws + off); off += (size_t)Bq   * Ddim * sizeof(_Float16);
    _Float16* mn = (_Float16*)(ws + off); off += (size_t)Nmem * Ddim * sizeof(_Float16);
    off = (off + 255) & ~(size_t)255;
    float* pV = (float*)(ws + off);       off += (size_t)Bq * NCHUNK * TOPK * sizeof(float);
    int*   pI = (int*)(ws + off);         off += (size_t)Bq * NCHUNK * TOPK * sizeof(int);

    normalize_rows_f16<<<Bq,   256, 0, stream>>>(query, qn, Ddim);
    normalize_rows_f16<<<Nmem, 256, 0, stream>>>(memf,  mn, Ddim);

    dim3 g2(Bq / 32, NCHUNK);
    sim_topk_kernel<<<g2, 256, 0, stream>>>(qn, mn, pV, pI);

    finalize_kernel<<<(Bq + 255) / 256, 256, 0, stream>>>(pV, pI, memev, modev,
                                                          (float*)d_out);
}
